// iNGPDW_6133213299476
// MI455X (gfx1250) — matrix operand order repaired and checked by reading, not compiled, not run on hardware
//
#include <hip/hip_runtime.h>
#include <hip/hip_bf16.h>
#include <math.h>

// ---------------------------------------------------------------------------
// Instant-NGP forward, fused: hash-grid encode + 4 WMMA MLP heads (CDNA5)
// v2: templated heads (full unroll), hoisted A/B fragments, branchless
//     hardware-exp transcendentals (v_exp_f32 co-executes with WMMA).
// ---------------------------------------------------------------------------

typedef __attribute__((ext_vector_type(16))) _Float16 v16h;
typedef __attribute__((ext_vector_type(8)))  float    v8f;

#define LDS_STRIDE 72        // 64 K-cols + 8 pad halves (bank-conflict pad)
#define WAVES_PER_BLOCK 4
#define PTS_PER_BLOCK 128    // 4 waves * 32 points

__constant__ float kRes[10] = {16.f, 34.f, 74.f, 161.f, 348.f,
                               752.f, 1625.f, 3511.f, 7584.f, 16384.f};

union FragU { v16h v; unsigned int u[8]; };

__device__ __forceinline__ void wave_lds_fence() {
  // DS ops from one wave are processed in order; wait for all outstanding
  // LDS stores so other lanes of this wave can read the data. No workgroup
  // barrier needed: each wave owns a private LDS slice.
  asm volatile("s_wait_dscnt 0x0" ::: "memory");
}

__device__ __forceinline__ v8f wmma16(v16h a, v16h b, v8f c) {
  // v_wmma_f32_16x16x32_f16: D = A(16x32 f16) * B(32x16 f16) + C(16x16 f32)
  return __builtin_amdgcn_wmma_f32_16x16x32_f16(
      false, a, false, b, (short)0, c, false, false);
}

// A fragment (16x32, f16) from LDS row-major [m][LDS_STRIDE].
// ISA 7.12.2 16-bit A layout: lanes 0-15 hold row m=lane, K slots
// {0..7,16..23}; lanes 16-31 hold row m=lane-16, K slots {8..15,24..31}.
__device__ __forceinline__ v16h load_frag_a(const _Float16* tile, int lane, int kBase) {
  const int m    = lane & 15;
  const int koff = (lane < 16) ? 0 : 8;
  const unsigned int* row = (const unsigned int*)(tile + m * LDS_STRIDE);
  const int b0 = (kBase + koff) >> 1;        // dword index of K pair
  const int b1 = (kBase + 16 + koff) >> 1;
  FragU f;
#pragma unroll
  for (int j = 0; j < 4; ++j) {
    f.u[j]     = row[b0 + j];
    f.u[4 + j] = row[b1 + j];
  }
  return f.v;
}

// B fragment (32x16, f16) from global W^T stored [n][64] f16.
// 16-bit B layout: lane holds column n=lane&15; lanes 0-15 K slots
// {0..7,16..23}, lanes 16-31 K slots {8..15,24..31} (matches A).
__device__ __forceinline__ v16h load_frag_b(const _Float16* wt, int lane, int kBase) {
  const int n  = lane & 15;
  const int koff = (lane < 16) ? 0 : 8;
  const unsigned int* row = (const unsigned int*)(wt + n * 64);
  const int b0 = (kBase + koff) >> 1;
  const int b1 = (kBase + 16 + koff) >> 1;
  FragU f;
#pragma unroll
  for (int j = 0; j < 4; ++j) {
    f.u[j]     = row[b0 + j];
    f.u[4 + j] = row[b1 + j];
  }
  return f.v;
}

// Branchless SELU on the hardware exp path (v_exp_f32 is a TRANS op that
// co-executes with the XDL/WMMA pipe; no EXEC-mask branches).
__device__ __forceinline__ float selu_f(float v) {
  const float a = 1.6732632423543772f;
  const float s = 1.0507009873554805f;
  float neg = a * (__expf(v) - 1.0f);
  return s * (v > 0.f ? v : neg);     // -> v_cndmask, no branch
}

// One hidden layer: out[32][64] = selu(in[32][64] @ W(64x64) + b), staged f16.
// A frags hoisted (shared over N-tiles), B frags shared over both M-tiles.
__device__ __forceinline__ void hidden_layer(const _Float16* inBuf, _Float16* outBuf,
                                             const _Float16* wt, const float* bias,
                                             int lane) {
  const int nlo = lane & 15;
  const int hi  = (lane >= 16) ? 8 : 0;
  const v16h a00 = load_frag_a(inBuf,                   lane, 0);
  const v16h a01 = load_frag_a(inBuf,                   lane, 32);
  const v16h a10 = load_frag_a(inBuf + 16 * LDS_STRIDE, lane, 0);
  const v16h a11 = load_frag_a(inBuf + 16 * LDS_STRIDE, lane, 32);
#pragma unroll
  for (int nt = 0; nt < 4; ++nt) {
    const v16h b0 = load_frag_b(wt + nt * 16 * 64, lane, 0);
    const v16h b1 = load_frag_b(wt + nt * 16 * 64, lane, 32);
    v8f acc0 = {};
    v8f acc1 = {};
    acc0 = wmma16(a00, b0, acc0);
    acc1 = wmma16(a10, b0, acc1);
    acc0 = wmma16(a01, b1, acc0);
    acc1 = wmma16(a11, b1, acc1);
    const float bv = bias[nt * 16 + nlo];
    const int n = nt * 16 + nlo;
#pragma unroll
    for (int v = 0; v < 8; ++v) {
      outBuf[(v + hi) * LDS_STRIDE + n]      = (_Float16)selu_f(acc0[v] + bv);
      outBuf[(16 + v + hi) * LDS_STRIDE + n] = (_Float16)selu_f(acc1[v] + bv);
    }
  }
  wave_lds_fence();
}

// Final layer (single 16-col N tile) + head-specific epilogue to d_out.
template <int HEAD, int OUTD>
__device__ __forceinline__ void final_layer(const _Float16* inBuf, const _Float16* wt,
                                            const float* bias,
                                            float* __restrict__ out, size_t Npts,
                                            size_t pBase, int lane) {
  const int n  = lane & 15;
  const int hi = (lane >= 16) ? 8 : 0;
  const v16h b0 = load_frag_b(wt, lane, 0);
  const v16h b1 = load_frag_b(wt, lane, 32);
  const float bv = (n < OUTD) ? bias[n] : 0.f;
#pragma unroll
  for (int mt = 0; mt < 2; ++mt) {
    v8f acc = {};
    acc = wmma16(load_frag_a(inBuf + mt * 16 * LDS_STRIDE, lane, 0),  b0, acc);
    acc = wmma16(load_frag_a(inBuf + mt * 16 * LDS_STRIDE, lane, 32), b1, acc);
#pragma unroll
    for (int v = 0; v < 8; ++v) {
      const float  val = acc[v] + bv;
      const size_t p   = pBase + (size_t)(mt * 16 + v + hi);
      if (HEAD == 0) {               // density = exp(min(sigma - 4, 15))
        if (n == 0) out[p] = __expf(fminf(val - 4.0f, 15.0f));
      } else if (HEAD == 1) {        // rgb = val + 0.5
        if (n < 3) out[Npts + p * 3 + n] = val + 0.5f;
      } else if (HEAD == 2) {        // grd = tanh(val)/sqrt(3), exp-based tanh
        if (n < 3) {
          const float e = __expf(2.0f * val);
          const float t = 1.0f - 2.0f * __builtin_amdgcn_rcpf(e + 1.0f);
          out[4 * Npts + p * 3 + n] = t * 0.57735026918962576f;
        }
      } else {                       // sh, rounded through f16
        out[7 * Npts + p * 16 + n] = (float)(_Float16)val;
      }
    }
  }
}

template <int HEAD, int OUTD>
__device__ __forceinline__ void run_head(const _Float16* enc, _Float16* act0,
                                         _Float16* act1, const _Float16* hw,
                                         const float* b1, const float* b2,
                                         const float* b3, float* __restrict__ out,
                                         size_t Npts, size_t pBase, int lane) {
  hidden_layer(enc,  act0, hw,        b1, lane);
  hidden_layer(act0, act1, hw + 4096, b2, lane);
  final_layer<HEAD, OUTD>(act1, hw + 8192, b3, out, Npts, pBase, lane);
}

// ---------------------------------------------------------------------------
// Weight prep: f32 [K][N] -> f16 transposed-padded [N][64] per head in d_ws.
// Per-head layout (f16): W1t [64][64] @0, W2t [64][64] @4096, W3t [16][64] @8192.
// ---------------------------------------------------------------------------
__global__ void prep_weights(const float* dW1, const float* dW2, const float* dW3,
                             const float* cW1, const float* cW2, const float* cW3,
                             const float* gW1, const float* gW2, const float* gW3,
                             const float* sW1, const float* sW2, const float* sW3,
                             _Float16* __restrict__ wsW) {
  const int PER_HEAD = 9216;
  int tid = blockIdx.x * blockDim.x + threadIdx.x;
  if (tid >= 4 * PER_HEAD) return;
  const int head = tid / PER_HEAD;
  const int r    = tid - head * PER_HEAD;
  const float* W1 = (head == 0) ? dW1 : (head == 1) ? cW1 : (head == 2) ? gW1 : sW1;
  const float* W2 = (head == 0) ? dW2 : (head == 1) ? cW2 : (head == 2) ? gW2 : sW2;
  const float* W3 = (head == 0) ? dW3 : (head == 1) ? cW3 : (head == 2) ? gW3 : sW3;
  const int outDim = (head == 0) ? 1 : (head == 3) ? 16 : 3;
  float v;
  if (r < 4096) {                      // W1: [40][64] -> [n][k<64]
    int n = r >> 6, k = r & 63;
    v = (k < 40) ? W1[k * 64 + n] : 0.f;
  } else if (r < 8192) {               // W2: [64][64]
    int q = r - 4096; int n = q >> 6, k = q & 63;
    v = W2[k * 64 + n];
  } else {                             // W3: [64][outDim] -> [16 rows pad]
    int q = r - 8192; int n = q >> 6, k = q & 63;
    v = (n < outDim) ? W3[k * outDim + n] : 0.f;
  }
  wsW[tid] = (_Float16)v;
}

// ---------------------------------------------------------------------------
// Fused main kernel: encode -> 4 heads of 3-layer WMMA MLP.
// 128 threads = 4 waves; each wave owns 32 points + private LDS slice.
// N (=524288) is divisible by 128 -> no tail guard -> EXEC all-1s for WMMA.
// ---------------------------------------------------------------------------
__global__ void __launch_bounds__(128) ingp_fused(
    const float* __restrict__ x, const float* __restrict__ cr,
    const float* __restrict__ tables, const _Float16* __restrict__ wsW,
    const float* __restrict__ db1, const float* __restrict__ db2, const float* __restrict__ db3,
    const float* __restrict__ cb1, const float* __restrict__ cb2, const float* __restrict__ cb3,
    const float* __restrict__ gb1, const float* __restrict__ gb2, const float* __restrict__ gb3,
    const float* __restrict__ sb1, const float* __restrict__ sb2, const float* __restrict__ sb3,
    float* __restrict__ out, size_t Npts) {
  __shared__ _Float16 s_enc [WAVES_PER_BLOCK * 32 * LDS_STRIDE];
  __shared__ _Float16 s_act0[WAVES_PER_BLOCK * 32 * LDS_STRIDE];
  __shared__ _Float16 s_act1[WAVES_PER_BLOCK * 32 * LDS_STRIDE];

  const int lane = threadIdx.x & 31;
  const int wave = threadIdx.x >> 5;
  const size_t pBase = (size_t)blockIdx.x * PTS_PER_BLOCK + (size_t)wave * 32;
  const size_t p = pBase + (size_t)lane;

  _Float16* enc  = s_enc  + wave * 32 * LDS_STRIDE;
  _Float16* act0 = s_act0 + wave * 32 * LDS_STRIDE;
  _Float16* act1 = s_act1 + wave * 32 * LDS_STRIDE;

  // ---- hash-grid encoding: one point per lane -> f16 row [K=64 padded] ----
  {
    const float px0 = x[3 * p + 0], px1 = x[3 * p + 1], px2 = x[3 * p + 2];
    const float crv = cr[p] * 0.5f;                 // cr * SCALE_MULTI
    const float4* tab = (const float4*)tables;      // [L][65536][4] f32
    _Float16* encRow = enc + lane * LDS_STRIDE;
    unsigned int* zrow = (unsigned int*)(encRow + 40);  // zero K=40..63
#pragma unroll
    for (int j = 0; j < 12; ++j) zrow[j] = 0u;
#pragma unroll
    for (int l = 0; l < 10; ++l) {
      const float res = kRes[l];
      const float fx = px0 * res, fy = px1 * res, fz = px2 * res;
      const float f0x = floorf(fx), f0y = floorf(fy), f0z = floorf(fz);
      const unsigned int ix = (unsigned int)f0x;
      const unsigned int iy = (unsigned int)f0y;
      const unsigned int iz = (unsigned int)f0z;
      const float wx = fx - f0x, wy = fy - f0y, wz = fz - f0z;
      float a0 = 0.f, a1 = 0.f, a2 = 0.f, a3 = 0.f;
#pragma unroll
      for (int c = 0; c < 8; ++c) {
        const unsigned int cx = ix + ((c >> 2) & 1);
        const unsigned int cy = iy + ((c >> 1) & 1);
        const unsigned int cz = iz + (c & 1);
        const unsigned int hsh = cx ^ (cy * 2654435761u) ^ (cz * 805459861u);
        const unsigned int idx = hsh & 65535u;
        const float4 f = tab[(size_t)l * 65536 + idx];
        const float w = ((c & 4) ? wx : 1.f - wx) *
                        ((c & 2) ? wy : 1.f - wy) *
                        ((c & 1) ? wz : 1.f - wz);
        a0 = fmaf(w, f.x, a0); a1 = fmaf(w, f.y, a1);
        a2 = fmaf(w, f.z, a2); a3 = fmaf(w, f.w, a3);
      }
      // per-level erf scaling; l=0 -> erf(1e6)=1
      const float arg = fmaxf(8.0f * (float)l * crv, 1e-12f);
      const float s = erff(rsqrtf(arg));
      encRow[0 * 10 + l] = (_Float16)(a0 * s);   // K = d*L + l (reshape order)
      encRow[1 * 10 + l] = (_Float16)(a1 * s);
      encRow[2 * 10 + l] = (_Float16)(a2 * s);
      encRow[3 * 10 + l] = (_Float16)(a3 * s);
    }
  }
  wave_lds_fence();

  // ---- 4 MLP heads on WMMA (fully specialized / unrolled) ----
  run_head<0, 1>(enc, act0, act1, wsW + 0 * 9216, db1, db2, db3, out, Npts, pBase, lane);
  run_head<1, 3>(enc, act0, act1, wsW + 1 * 9216, cb1, cb2, cb3, out, Npts, pBase, lane);
  run_head<2, 3>(enc, act0, act1, wsW + 2 * 9216, gb1, gb2, gb3, out, Npts, pBase, lane);
  run_head<3, 16>(enc, act0, act1, wsW + 3 * 9216, sb1, sb2, sb3, out, Npts, pBase, lane);
}

// ---------------------------------------------------------------------------
// Launch. Inputs (setup_inputs order): x, cr, tables, then per head
// {W1,b1,W2,b2,W3,b3} for d, c, g, s. Output: [density N | rgb 3N | grd 3N |
// sh 16N] flat. d_ws needs 4*9216*2 = 73,728 bytes for f16 weights.
// ---------------------------------------------------------------------------
extern "C" void kernel_launch(void* const* d_in, const int* in_sizes, int n_in,
                              void* d_out, int out_size, void* d_ws, size_t ws_size,
                              hipStream_t stream) {
  const float* x      = (const float*)d_in[0];
  const float* cr     = (const float*)d_in[1];
  const float* tables = (const float*)d_in[2];
  const float* W[24];
  for (int i = 0; i < 24; ++i) W[i] = (const float*)d_in[3 + i];
  _Float16* wsW = (_Float16*)d_ws;
  const size_t Npts = (size_t)in_sizes[1];   // cr has N elements

  prep_weights<<<(4 * 9216 + 255) / 256, 256, 0, stream>>>(
      W[0],  W[2],  W[4],    // d:  W1 W2 W3
      W[6],  W[8],  W[10],   // c
      W[12], W[14], W[16],   // g
      W[18], W[20], W[22],   // s
      wsW);

  dim3 grid((unsigned)(Npts / PTS_PER_BLOCK));
  ingp_fused<<<grid, PTS_PER_BLOCK, 0, stream>>>(
      x, cr, tables, wsW,
      W[1],  W[3],  W[5],    // d biases
      W[7],  W[9],  W[11],   // c
      W[13], W[15], W[17],   // g
      W[19], W[21], W[23],   // s
      (float*)d_out, Npts);
}
